// ViewAlignedFeatureTransformer_7301444403336
// MI455X (gfx1250) — matrix operand order repaired and checked by reading, not compiled, not run on hardware
//
#include <hip/hip_runtime.h>
#include <hip/hip_bf16.h>
#include <math.h>

// ---------------------------------------------------------------------------
// MI455X (gfx1250, wave32) implementation of the dual-attention block.
// All large GEMMs use v_wmma_f32_16x16x32_bf16 with f32 accumulation, with
// double-buffered GLOBAL_LOAD_ASYNC_TO_LDS_B128 staging (ASYNCcnt pipeline).
// ---------------------------------------------------------------------------

using u16 = unsigned short;
using u32 = unsigned int;

typedef __attribute__((ext_vector_type(16))) __bf16 v16bf;
typedef __attribute__((ext_vector_type(8)))  float  v8f;

__device__ __forceinline__ u16 f2bf(float f) {
  u32 u = __float_as_uint(f);
  u32 r = 0x7fffu + ((u >> 16) & 1u);   // round-to-nearest-even
  return (u16)((u + r) >> 16);
}
__device__ __forceinline__ float bf2f(u16 h) {
  return __uint_as_float(((u32)h) << 16);
}
__device__ __forceinline__ float wave_sum(float v) {
#pragma unroll
  for (int m = 16; m > 0; m >>= 1) v += __shfl_xor(v, m, 32);
  return v;
}
__device__ __forceinline__ float wave_max(float v) {
#pragma unroll
  for (int m = 16; m > 0; m >>= 1) v = fmaxf(v, __shfl_xor(v, m, 32));
  return v;
}

// LDS byte offset of a __shared__ object (ptrtoint of an addrspace(3) ptr).
__device__ __forceinline__ u32 lds_off(const void* p) {
  return (u32)(unsigned long long)(__attribute__((address_space(3))) const void*)p;
}

// Async DMA: 16B per lane, global -> LDS, tracked by ASYNCcnt.
__device__ __forceinline__ void async_copy16(u32 lds_byte_off, const void* gaddr) {
  asm volatile("global_load_async_to_lds_b128 %0, %1, off"
               :: "v"(lds_byte_off), "v"(gaddr)
               : "memory");
}
__device__ __forceinline__ void wait_async0() {
  asm volatile("s_wait_asynccnt 0x0" ::: "memory");
}

// ---------------------------------------------------------------------------
// WMMA bf16 GEMM:  C[M,N] = A[M,K] @ Bt[N,K]^T  (Bt = B stored transposed)
// Block tile 128x128, K-step 32, 8 waves (4x2), 2x4 fragments per wave.
// PIPE=true : async-to-LDS double-buffered staging; REQUIRES M%128==0 &&
//             N%128==0 (no guards anywhere).
// PIPE=false: synchronous guarded staging (used for PV where N=64).
// K must be a multiple of 32 (true for all GEMMs here: 64/512/1024/2048).
// EPI: 0 = f32 out (+bias, *scale, batched)   1 = f32 out + bias + residual
//      2 = bf16 out (+bias)                   3 = final NCHW store + x resid
// ---------------------------------------------------------------------------
#define BM 128
#define BN 128
#define BKT 32
#define LDT 40   // padded LDS row stride in halfwords (80B, 16B aligned)

union FragCast { uint4 q[2]; v16bf v; };

template <int EPI, bool PIPE>
__global__ __launch_bounds__(256) void gemm_bf16(
    const u16* __restrict__ A, long long sA,
    const u16* __restrict__ Bt, long long sB,
    void* __restrict__ Cvp, long long sC,
    const float* __restrict__ bias,
    const float* __restrict__ resid,
    const float* __restrict__ xin,
    int M, int N, int K, float scale)
{
  __shared__ u16 As[2][BM][LDT];
  __shared__ u16 Bs[2][BN][LDT];

  const int tid  = threadIdx.x;
  const int lane = tid & 31;
  const int wid  = tid >> 5;
  const int hi   = lane >> 4;
  const int ln   = lane & 15;
  const int bm   = blockIdx.y * BM;
  const int bn   = blockIdx.x * BN;
  const int z    = blockIdx.z;

  A  += (long long)z * sA;
  Bt += (long long)z * sB;

  const int wm = (wid >> 1) * 32;   // wave tile: 32 rows x 64 cols
  const int wn = (wid & 1) * 64;

  v8f acc[2][4] = {};

  // Issue async staging of one 128x32 A-tile + 128x32 B-tile (4 x B128/lane).
  auto stage_async = [&](int buf, int k0) {
#pragma unroll
    for (int r = 0; r < 2; ++r) {
      int vId = tid + r * 256;          // 0..511
      int row = vId >> 2;
      int col = (vId & 3) << 3;
      async_copy16(lds_off(&As[buf][row][col]),
                   A + (size_t)(bm + row) * K + (k0 + col));
      async_copy16(lds_off(&Bs[buf][row][col]),
                   Bt + (size_t)(bn + row) * K + (k0 + col));
    }
  };

  const int nk = K / BKT;
  if (PIPE) {
    stage_async(0, 0);
    wait_async0();
    __syncthreads();
  }

  for (int ki = 0; ki < nk; ++ki) {
    const int k0 = ki * BKT;
    const int cur = PIPE ? (ki & 1) : 0;

    if (PIPE) {
      if (ki + 1 < nk) stage_async((ki + 1) & 1, k0 + BKT);  // overlap DMA
    } else {
      // guarded synchronous staging into buffer 0
#pragma unroll
      for (int r = 0; r < 2; ++r) {
        int vId = tid + r * 256;
        int row = vId >> 2;
        int col = (vId & 3) << 3;
        uint4 da = make_uint4(0u, 0u, 0u, 0u);
        int gr = bm + row;
        if (gr < M) da = *(const uint4*)(A + (size_t)gr * K + (k0 + col));
        *(uint4*)&As[0][row][col] = da;
        uint4 db = make_uint4(0u, 0u, 0u, 0u);
        int gn = bn + row;
        if (gn < N) db = *(const uint4*)(Bt + (size_t)gn * K + (k0 + col));
        *(uint4*)&Bs[0][row][col] = db;
      }
      __syncthreads();
    }

    // Fragment gather per ISA 7.12.2 (16-bit A 16x32, B 32x16 layouts).
    v16bf af[2], bfr[4];
#pragma unroll
    for (int mi = 0; mi < 2; ++mi) {
      FragCast fc;
      int r = wm + mi * 16 + ln;
      fc.q[0] = *(const uint4*)&As[cur][r][hi * 8];        // K = hi*8 .. +7
      fc.q[1] = *(const uint4*)&As[cur][r][16 + hi * 8];   // K = 16+hi*8 ..
      af[mi] = fc.v;
    }
#pragma unroll
    for (int ni = 0; ni < 4; ++ni) {
      FragCast fc;
      int n = wn + ni * 16 + ln;
      fc.q[0] = *(const uint4*)&Bs[cur][n][hi * 8];        // K = hi*8 .. +7
      fc.q[1] = *(const uint4*)&Bs[cur][n][16 + hi * 8];   // K = 16+hi*8 ..
      bfr[ni] = fc.v;
    }

#pragma unroll
    for (int mi = 0; mi < 2; ++mi)
#pragma unroll
      for (int ni = 0; ni < 4; ++ni)
        acc[mi][ni] = __builtin_amdgcn_wmma_f32_16x16x32_bf16(
            false, af[mi], false, bfr[ni], (short)0, acc[mi][ni], false, false);

    if (PIPE) wait_async0();   // next tile's DMA landed (this wave's share)
    __syncthreads();           // all waves done reading cur / writing next
  }

  // Epilogue (C/D layout: VGPR e -> row e + hi*8, col = ln within 16x16 tile)
#pragma unroll
  for (int mi = 0; mi < 2; ++mi) {
#pragma unroll
    for (int ni = 0; ni < 4; ++ni) {
      int col = bn + wn + ni * 16 + ln;
      if (!PIPE && col >= N) continue;
      float bv = bias ? bias[col] : 0.0f;
      int rbase = bm + wm + mi * 16 + hi * 8;
#pragma unroll
      for (int e = 0; e < 8; ++e) {
        int row = rbase + e;
        if (!PIPE && row >= M) continue;
        float v = acc[mi][ni][e] * scale + bv;
        size_t idx = (size_t)row * N + col;
        if (EPI == 0) {
          ((float*)Cvp)[(size_t)z * sC + idx] = v;
        } else if (EPI == 1) {
          ((float*)Cvp)[idx] = v + resid[idx];
        } else if (EPI == 2) {
          ((u16*)Cvp)[idx] = f2bf(v);
        } else {
          int b  = row >> 10;           // token -> (batch, hw), HW = 1024
          int hw = row & 1023;
          size_t oidx = ((size_t)b * 512 + col) * 1024 + hw;
          ((float*)Cvp)[oidx] = v + xin[oidx];
        }
      }
    }
  }
}

// ---------------------------------------------------------------------------
// Weight convert + transpose: W[K,N] f32 -> Wt[N,K] bf16
// ---------------------------------------------------------------------------
__global__ __launch_bounds__(256) void wconvT_kernel(
    const float* __restrict__ W, u16* __restrict__ Wt, int K, int N)
{
  size_t idx = (size_t)blockIdx.x * 256 + threadIdx.x;
  if (idx >= (size_t)K * N) return;
  int n = (int)(idx % N);
  int k = (int)(idx / N);
  Wt[(size_t)n * K + k] = f2bf(W[idx]);
}

__global__ __launch_bounds__(256) void convert_bf16_kernel(
    const float* __restrict__ in, u16* __restrict__ out, long long n)
{
  size_t idx = (size_t)blockIdx.x * 256 + threadIdx.x;
  if (idx < (size_t)n) out[idx] = f2bf(in[idx]);
}

// ---------------------------------------------------------------------------
// GroupNorm(32 groups, eps 1e-6) fused with NCHW -> (token, C) bf16 permute.
// One block per (batch, group): 16 channels x 1024 pixels.
// ---------------------------------------------------------------------------
__global__ __launch_bounds__(256) void groupnorm_kernel(
    const float* __restrict__ x, const float* __restrict__ gw,
    const float* __restrict__ gb, u16* __restrict__ out)
{
  const int b = blockIdx.x >> 5;
  const int g = blockIdx.x & 31;
  const int CH = 16, HW = 1024, C = 512;
  const float* base = x + ((size_t)b * C + (size_t)g * CH) * HW;

  float s = 0.f, s2 = 0.f;
  for (int i = threadIdx.x; i < CH * HW; i += 256) {
    float v = base[i]; s += v; s2 += v * v;
  }
  __shared__ float red0[8], red1[8];
  s = wave_sum(s); s2 = wave_sum(s2);
  int wid = threadIdx.x >> 5, lane = threadIdx.x & 31;
  if (lane == 0) { red0[wid] = s; red1[wid] = s2; }
  __syncthreads();
  float ts = 0.f, ts2 = 0.f;
#pragma unroll
  for (int i = 0; i < 8; ++i) { ts += red0[i]; ts2 += red1[i]; }
  const float inv_n = 1.0f / (CH * HW);
  float mu = ts * inv_n;
  float var = ts2 * inv_n - mu * mu;
  float rs = rsqrtf(var + 1e-6f);

  for (int i = threadIdx.x; i < CH * HW; i += 256) {
    int ci = i >> 10;            // i / HW
    int hw = i & 1023;
    int c = g * CH + ci;
    float v = (base[i] - mu) * rs * gw[c] + gb[c];
    out[((size_t)b * HW + hw) * C + c] = f2bf(v);
  }
}

// ---------------------------------------------------------------------------
// LayerNorm over 512 channels, f32 in -> bf16 out. One wave per token.
// ---------------------------------------------------------------------------
__global__ __launch_bounds__(256) void layernorm_bf16_kernel(
    const float* __restrict__ x, const float* __restrict__ w,
    const float* __restrict__ b, u16* __restrict__ out, int M)
{
  int row = blockIdx.x * 8 + (threadIdx.x >> 5);
  int lane = threadIdx.x & 31;
  if (row >= M) return;
  const float* xr = x + (size_t)row * 512;
  float vals[16];
  float s = 0.f;
#pragma unroll
  for (int i = 0; i < 16; ++i) { vals[i] = xr[lane + i * 32]; s += vals[i]; }
  s = wave_sum(s);
  float mu = s * (1.0f / 512.0f);
  float s2 = 0.f;
#pragma unroll
  for (int i = 0; i < 16; ++i) { float d = vals[i] - mu; s2 += d * d; }
  s2 = wave_sum(s2);
  float rs = rsqrtf(s2 * (1.0f / 512.0f) + 1e-5f);
  u16* orow = out + (size_t)row * 512;
#pragma unroll
  for (int i = 0; i < 16; ++i) {
    int c = lane + i * 32;
    orow[c] = f2bf((vals[i] - mu) * rs * w[c] + b[c]);
  }
}

// ---------------------------------------------------------------------------
// Split heads: q/k f32 [8192,512] -> [B*H, S, 64] bf16; v -> [B*H, 64, S]
// (V transposed so PV's B-fragment LDS reads are contiguous).
// ---------------------------------------------------------------------------
__global__ __launch_bounds__(256) void repack_qkv_kernel(
    const float* __restrict__ q, const float* __restrict__ k,
    const float* __restrict__ v, u16* __restrict__ qh,
    u16* __restrict__ kh, u16* __restrict__ vT)
{
  size_t idx = (size_t)blockIdx.x * 256 + threadIdx.x;
  if (idx >= (size_t)8192 * 512) return;
  int col = (int)(idx & 511);
  size_t tok = idx >> 9;
  int b = (int)(tok >> 10);
  int s = (int)(tok & 1023);
  int h = col >> 6, dh = col & 63;
  size_t bh = (size_t)b * 8 + h;
  qh[(bh * 1024 + s) * 64 + dh] = f2bf(q[idx]);
  kh[(bh * 1024 + s) * 64 + dh] = f2bf(k[idx]);
  vT[(bh * 64 + dh) * 1024 + s] = f2bf(v[idx]);
}

// Merge heads: [B*H, S, 64] f32 -> [token, 512] bf16
__global__ __launch_bounds__(256) void merge_heads_kernel(
    const float* __restrict__ ah, u16* __restrict__ out)
{
  size_t idx = (size_t)blockIdx.x * 256 + threadIdx.x;
  if (idx >= (size_t)8192 * 512) return;
  int col = (int)(idx & 511);
  size_t tok = idx >> 9;
  int b = (int)(tok >> 10);
  int s = (int)(tok & 1023);
  int h = col >> 6, dh = col & 63;
  out[idx] = f2bf(ah[(((size_t)b * 8 + h) * 1024 + s) * 64 + dh]);
}

// ---------------------------------------------------------------------------
// Row softmax over 1024 keys, f32 scores -> bf16 probs. One wave per row.
// ---------------------------------------------------------------------------
__global__ __launch_bounds__(256) void softmax_kernel(
    const float* __restrict__ s, u16* __restrict__ p, int rows)
{
  int row = blockIdx.x * 8 + (threadIdx.x >> 5);
  int lane = threadIdx.x & 31;
  if (row >= rows) return;
  const float* sr = s + (size_t)row * 1024;
  float vals[32];
  float mx = -3.0e38f;
#pragma unroll
  for (int i = 0; i < 32; ++i) { vals[i] = sr[lane + i * 32]; mx = fmaxf(mx, vals[i]); }
  mx = wave_max(mx);
  float sum = 0.f;
#pragma unroll
  for (int i = 0; i < 32; ++i) { vals[i] = __expf(vals[i] - mx); sum += vals[i]; }
  sum = wave_sum(sum);
  float inv = 1.0f / sum;
  u16* pr = p + (size_t)row * 1024;
#pragma unroll
  for (int i = 0; i < 32; ++i) pr[lane + i * 32] = f2bf(vals[i] * inv);
}

// ---------------------------------------------------------------------------
// Per-pixel cross attention: 1 query x 8 keys per (pixel, head).
// One wave per (pixel, head); each lane owns 2 of the 64 head dims.
// WMMA tiles (16x16) would be >90% padding at Sk=8, so this stays on VALU.
// ---------------------------------------------------------------------------
__global__ __launch_bounds__(256) void pixattn_kernel(
    const u16* __restrict__ q2, const u16* __restrict__ k2,
    const u16* __restrict__ v2, u16* __restrict__ out)
{
  int unit = blockIdx.x * 8 + (threadIdx.x >> 5);   // pixel*8 + head
  int lane = threadIdx.x & 31;
  int pix = unit >> 3;
  int h = unit & 7;
  int d0 = lane * 2;
  const u16* qr = q2 + (size_t)pix * 512 + h * 64;
  float q0 = bf2f(qr[d0]), q1 = bf2f(qr[d0 + 1]);
  float sc[8];
#pragma unroll
  for (int v = 0; v < 8; ++v) {
    const u16* kr = k2 + ((size_t)pix * 8 + v) * 512 + h * 64;
    float p = q0 * bf2f(kr[d0]) + q1 * bf2f(kr[d0 + 1]);
    sc[v] = wave_sum(p) * 0.125f;                   // DH^-0.5
  }
  float mx = sc[0];
#pragma unroll
  for (int v = 1; v < 8; ++v) mx = fmaxf(mx, sc[v]);
  float sum = 0.f;
#pragma unroll
  for (int v = 0; v < 8; ++v) { sc[v] = __expf(sc[v] - mx); sum += sc[v]; }
  float inv = 1.0f / sum;
  float o0 = 0.f, o1 = 0.f;
#pragma unroll
  for (int v = 0; v < 8; ++v) {
    const u16* vr = v2 + ((size_t)pix * 8 + v) * 512 + h * 64;
    float p = sc[v] * inv;
    o0 += p * bf2f(vr[d0]);
    o1 += p * bf2f(vr[d0 + 1]);
  }
  u16* orow = out + (size_t)pix * 512 + h * 64;
  orow[d0] = f2bf(o0);
  orow[d0 + 1] = f2bf(o1);
}

// ---------------------------------------------------------------------------
// GEGLU: h[8192,4096] f32 -> val * gelu_exact(gate) -> bf16 [8192,2048]
// ---------------------------------------------------------------------------
__global__ __launch_bounds__(256) void geglu_kernel(
    const float* __restrict__ hbuf, u16* __restrict__ out)
{
  size_t idx = (size_t)blockIdx.x * 256 + threadIdx.x;
  if (idx >= (size_t)8192 * 2048) return;
  size_t m = idx >> 11;
  int j = (int)(idx & 2047);
  float val = hbuf[m * 4096 + j];
  float g = hbuf[m * 4096 + 2048 + j];
  float ge = 0.5f * g * (1.0f + erff(g * 0.70710678118654752f));
  out[idx] = f2bf(val * ge);
}

// ---------------------------------------------------------------------------
// Host launcher
// ---------------------------------------------------------------------------
extern "C" void kernel_launch(void* const* d_in, const int* in_sizes, int n_in,
                              void* d_out, int out_size, void* d_ws, size_t ws_size,
                              hipStream_t stream)
{
  (void)in_sizes; (void)n_in; (void)out_size; (void)ws_size;
  const float* x       = (const float*)d_in[0];
  const float* context = (const float*)d_in[1];
  const float* gn_w = (const float*)d_in[2];
  const float* gn_b = (const float*)d_in[3];
  const float* pin_w = (const float*)d_in[4];
  const float* pin_b = (const float*)d_in[5];
  const float* n1_w = (const float*)d_in[6];
  const float* n1_b = (const float*)d_in[7];
  const float* n2_w = (const float*)d_in[8];
  const float* n2_b = (const float*)d_in[9];
  const float* n3_w = (const float*)d_in[10];
  const float* n3_b = (const float*)d_in[11];
  const float* a1_q = (const float*)d_in[12];
  const float* a1_k = (const float*)d_in[13];
  const float* a1_v = (const float*)d_in[14];
  const float* a1_o = (const float*)d_in[15];
  const float* a1_ob = (const float*)d_in[16];
  const float* a2_q = (const float*)d_in[17];
  const float* a2_k = (const float*)d_in[18];
  const float* a2_v = (const float*)d_in[19];
  const float* a2_o = (const float*)d_in[20];
  const float* a2_ob = (const float*)d_in[21];
  const float* ff1_w = (const float*)d_in[22];
  const float* ff1_b = (const float*)d_in[23];
  const float* ff2_w = (const float*)d_in[24];
  const float* ff2_b = (const float*)d_in[25];
  const float* pout_w = (const float*)d_in[26];
  const float* pout_b = (const float*)d_in[27];
  float* out = (float*)d_out;

  // ---- workspace layout (256B aligned) ----
  char* ws = (char*)d_ws;
  size_t off = 0;
  auto alloc = [&](size_t bytes) -> char* {
    off = (off + 255) & ~(size_t)255;
    char* p = ws + off;
    off += bytes;
    return p;
  };
  const size_t MT = 8192;                 // tokens
  u16* pinT  = (u16*)alloc(512 * 512 * 2);
  u16* a1qT  = (u16*)alloc(512 * 512 * 2);
  u16* a1kT  = (u16*)alloc(512 * 512 * 2);
  u16* a1vT  = (u16*)alloc(512 * 512 * 2);
  u16* a1oT  = (u16*)alloc(512 * 512 * 2);
  u16* a2qT  = (u16*)alloc(512 * 512 * 2);
  u16* a2kT  = (u16*)alloc(512 * 512 * 2);
  u16* a2vT  = (u16*)alloc(512 * 512 * 2);
  u16* a2oT  = (u16*)alloc(512 * 512 * 2);
  u16* ff1T  = (u16*)alloc((size_t)4096 * 512 * 2);
  u16* ff2T  = (u16*)alloc((size_t)512 * 2048 * 2);
  u16* poutT = (u16*)alloc(512 * 512 * 2);
  float* t    = (float*)alloc(MT * 512 * 4);
  u16* bufA   = (u16*)alloc(MT * 512 * 2);
  u16* bufB   = (u16*)alloc(MT * 512 * 2);
  float* qf   = (float*)alloc(MT * 512 * 4);
  float* kf   = (float*)alloc(MT * 512 * 4);
  float* vf   = (float*)alloc(MT * 512 * 4);
  u16* qh     = (u16*)alloc(MT * 512 * 2);
  u16* kh     = (u16*)alloc(MT * 512 * 2);
  u16* vTb    = (u16*)alloc(MT * 512 * 2);
  float* attnh = (float*)alloc((size_t)64 * 1024 * 64 * 4);
  u16* q2b    = (u16*)alloc(MT * 512 * 2);
  u16* gbuf   = (u16*)alloc(MT * 2048 * 2);
  // time-shared big region (384 MB peak):
  char* big = alloc((size_t)384 * 1024 * 1024);
  float* scores = (float*)big;                                // 256 MB (attn1)
  u16* probs    = (u16*)(big + (size_t)256 * 1024 * 1024);    // 128 MB (attn1)
  u16* ctxb     = (u16*)big;                                  //  64 MB (attn2)
  u16* k2b      = (u16*)(big + (size_t)64 * 1024 * 1024);     //  64 MB (attn2)
  u16* v2b      = (u16*)(big + (size_t)128 * 1024 * 1024);    //  64 MB (attn2)
  float* ffh    = (float*)big;                                // 128 MB (FF)

  dim3 blk(256);
  auto cdiv = [](long long a, long long b) { return (unsigned)((a + b - 1) / b); };
  auto gg = [&](int M, int N, int Z) { return dim3(cdiv(N, 128), cdiv(M, 128), Z); };

  // 0) convert + transpose all weights to bf16 [N,K]
  {
    struct WEnt { const float* w; u16* tt; int K, N; };
    WEnt wl[12] = {
      {pin_w, pinT, 512, 512},  {a1_q, a1qT, 512, 512}, {a1_k, a1kT, 512, 512},
      {a1_v, a1vT, 512, 512},   {a1_o, a1oT, 512, 512}, {a2_q, a2qT, 512, 512},
      {a2_k, a2kT, 512, 512},   {a2_v, a2vT, 512, 512}, {a2_o, a2oT, 512, 512},
      {ff1_w, ff1T, 512, 4096}, {ff2_w, ff2T, 2048, 512}, {pout_w, poutT, 512, 512},
    };
    for (int i = 0; i < 12; ++i) {
      long long n = (long long)wl[i].K * wl[i].N;
      wconvT_kernel<<<dim3(cdiv(n, 256)), blk, 0, stream>>>(wl[i].w, wl[i].tt,
                                                            wl[i].K, wl[i].N);
    }
  }

  // 1) GroupNorm + permute -> bufA (bf16 tokens x 512)
  groupnorm_kernel<<<dim3(256), blk, 0, stream>>>(x, gn_w, gn_b, bufA);

  // 2) proj_in: t = bufA @ pin_w + pin_b
  gemm_bf16<0, true><<<gg(8192, 512, 1), blk, 0, stream>>>(
      bufA, 0, pinT, 0, t, 0, pin_b, nullptr, nullptr, 8192, 512, 512, 1.0f);

  // 3) LN1 -> bufA
  layernorm_bf16_kernel<<<dim3(1024), blk, 0, stream>>>(t, n1_w, n1_b, bufA, 8192);

  // 4) Q/K/V projections (no bias)
  gemm_bf16<0, true><<<gg(8192, 512, 1), blk, 0, stream>>>(
      bufA, 0, a1qT, 0, qf, 0, nullptr, nullptr, nullptr, 8192, 512, 512, 1.0f);
  gemm_bf16<0, true><<<gg(8192, 512, 1), blk, 0, stream>>>(
      bufA, 0, a1kT, 0, kf, 0, nullptr, nullptr, nullptr, 8192, 512, 512, 1.0f);
  gemm_bf16<0, true><<<gg(8192, 512, 1), blk, 0, stream>>>(
      bufA, 0, a1vT, 0, vf, 0, nullptr, nullptr, nullptr, 8192, 512, 512, 1.0f);

  // 5) split heads (V transposed)
  repack_qkv_kernel<<<dim3(16384), blk, 0, stream>>>(qf, kf, vf, qh, kh, vTb);

  // 6) batched QK^T * 1/sqrt(64): 64 x [1024,1024,64]
  gemm_bf16<0, true><<<gg(1024, 1024, 64), blk, 0, stream>>>(
      qh, 1024 * 64, kh, 1024 * 64, scores, 1024 * 1024,
      nullptr, nullptr, nullptr, 1024, 1024, 64, 0.125f);

  // 7) softmax rows -> bf16 probs
  softmax_kernel<<<dim3(8192), blk, 0, stream>>>(scores, probs, 65536);

  // 8) batched PV: 64 x [1024,64,1024]  (N=64 -> guarded sync path)
  gemm_bf16<0, false><<<gg(1024, 64, 64), blk, 0, stream>>>(
      probs, 1024 * 1024, vTb, 64 * 1024, attnh, 1024 * 64,
      nullptr, nullptr, nullptr, 1024, 64, 1024, 1.0f);

  // 9) merge heads -> bufA
  merge_heads_kernel<<<dim3(16384), blk, 0, stream>>>(attnh, bufA);

  // 10) attn1 output proj + residual into t
  gemm_bf16<1, true><<<gg(8192, 512, 1), blk, 0, stream>>>(
      bufA, 0, a1oT, 0, t, 0, a1_ob, t, nullptr, 8192, 512, 512, 1.0f);

  // 11) LN2 -> bufA; q2 = bufA @ a2_q (bf16 out)
  layernorm_bf16_kernel<<<dim3(1024), blk, 0, stream>>>(t, n2_w, n2_b, bufA, 8192);
  gemm_bf16<2, true><<<gg(8192, 512, 1), blk, 0, stream>>>(
      bufA, 0, a2qT, 0, q2b, 0, nullptr, nullptr, nullptr, 8192, 512, 512, 1.0f);

  // 12) context -> bf16
  convert_bf16_kernel<<<dim3(cdiv((long long)65536 * 512, 256)), blk, 0, stream>>>(
      context, ctxb, (long long)65536 * 512);

  // 13) K2/V2 projections: [65536,512] @ [512,512]
  gemm_bf16<2, true><<<gg(65536, 512, 1), blk, 0, stream>>>(
      ctxb, 0, a2kT, 0, k2b, 0, nullptr, nullptr, nullptr, 65536, 512, 512, 1.0f);
  gemm_bf16<2, true><<<gg(65536, 512, 1), blk, 0, stream>>>(
      ctxb, 0, a2vT, 0, v2b, 0, nullptr, nullptr, nullptr, 65536, 512, 512, 1.0f);

  // 14) per-pixel 8-key attention -> bufB
  pixattn_kernel<<<dim3(8192), blk, 0, stream>>>(q2b, k2b, v2b, bufB);

  // 15) attn2 output proj + residual into t
  gemm_bf16<1, true><<<gg(8192, 512, 1), blk, 0, stream>>>(
      bufB, 0, a2oT, 0, t, 0, a2_ob, t, nullptr, 8192, 512, 512, 1.0f);

  // 16) LN3 -> bufA
  layernorm_bf16_kernel<<<dim3(1024), blk, 0, stream>>>(t, n3_w, n3_b, bufA, 8192);

  // 17) FF1: [8192,512] @ [512,4096] + bias -> ffh
  gemm_bf16<0, true><<<gg(8192, 4096, 1), blk, 0, stream>>>(
      bufA, 0, ff1T, 0, ffh, 0, ff1_b, nullptr, nullptr, 8192, 4096, 512, 1.0f);

  // 18) GEGLU -> gbuf (bf16)
  geglu_kernel<<<dim3(65536), blk, 0, stream>>>(ffh, gbuf);

  // 19) FF2 + residual into t
  gemm_bf16<1, true><<<gg(8192, 512, 1), blk, 0, stream>>>(
      gbuf, 0, ff2T, 0, t, 0, ff2_b, t, nullptr, 8192, 512, 2048, 1.0f);

  // 20) t -> bf16, then proj_out with NCHW permute + outer residual
  convert_bf16_kernel<<<dim3(16384), blk, 0, stream>>>(t, bufA, (long long)8192 * 512);
  gemm_bf16<3, true><<<gg(8192, 512, 1), blk, 0, stream>>>(
      bufA, 0, poutT, 0, out, 0, pout_b, nullptr, x, 8192, 512, 512, 1.0f);
}
